// WNGat_86174223827412
// MI455X (gfx1250) — matrix operand order repaired and checked by reading, not compiled, not run on hardware
//
#include <hip/hip_runtime.h>
#include <math.h>

// ---------------------------------------------------------------------------
// 3-layer GAT on MI455X (gfx1250, wave32).
//   layer: h = act @ W  via v_wmma_f32_16x16x32_f16 on pre-packed f16
//          fragments (fragment-order pack removes per-loop scalar loads and
//          cvts), per-head attention logits, segment softmax over dst
//          (global_atomic_max_u32 on a monotone float map + atomic_add_f32),
//          alpha-weighted scatter-add, bias + ELU.
// Dims: N=10000 nodes, F=256 feats, H=4 heads, C=64 ch/head, E=320000 edges
// (+N self loops).  Whole working set (~50 MB) is L2-resident (192 MB), so
// the kernel is issue/atomic bound, not HBM bound: keep the GEMM loop at
// 3x b128-pair loads + 2 wmma per K-step.
// ---------------------------------------------------------------------------

typedef __attribute__((ext_vector_type(16))) _Float16 v16h;
typedef __attribute__((ext_vector_type(8)))  float    v8f;

#define NNODES 10000
#define FDIM   256
#define HEADS  4
#define CH     64
#define NEDGES 320000
#define ETOT   (NEDGES + NNODES)   // with self loops
#define NEG_SLOPE 0.2f

#define MTILES ((NNODES + 15) / 16)   // 625
#define KSTEPS (FDIM / 32)            // 8
#define NTILES (FDIM / 16)            // 16
#define FRAG_HALVES 16                // halves per lane per fragment

// ------------------ pack A (activations) into fragment order ---------------
// Layout: [MTILES][KSTEPS][32 lanes][16 halves]  (32B per lane -> b128 pair)
// A-fragment (16x32 f16): row = lane&15, K(i) = ks*32 + (lane>>4)*8 + (i<8?i:i+8)
__global__ void gat_pack_a(const float* __restrict__ act,
                           _Float16* __restrict__ ap, int M) {
  int t = blockIdx.x * blockDim.x + threadIdx.x;   // mtile*KSTEPS*32 + ks*32 + lane
  if (t >= MTILES * KSTEPS * 32) return;
  int lane  = t & 31;
  int ks    = (t >> 5) & (KSTEPS - 1);
  int mtile = t >> 8;
  int m = mtile * 16 + (lane & 15);
  if (m >= M) m = M - 1;                           // clamp; store masked in GEMM
  const float* arow = act + (size_t)m * FDIM;
  int hsel = lane >> 4;
  v16h v;
#pragma unroll
  for (int i = 0; i < 16; ++i) {
    int k = ks * 32 + hsel * 8 + ((i < 8) ? i : (i + 8));
    v[i] = (_Float16)arow[k];
  }
  *(v16h*)(ap + (size_t)t * FRAG_HALVES) = v;
}

// ---------------------- pack W into fragment order -------------------------
// Layout: [NTILES][KSTEPS][32 lanes][16 halves]
// B-fragment (32x16 f16): col = lane&15, K(i) = ks*32 + (lane>>4)*8 + (i<8?i:i+8)
__global__ void gat_pack_w(const float* __restrict__ W,
                           _Float16* __restrict__ wp) {
  int t = blockIdx.x * blockDim.x + threadIdx.x;   // ntile*KSTEPS*32 + ks*32 + lane
  if (t >= NTILES * KSTEPS * 32) return;
  int lane  = t & 31;
  int ks    = (t >> 5) & (KSTEPS - 1);
  int ntile = t >> 8;
  int n = ntile * 16 + (lane & 15);
  int hsel = lane >> 4;
  v16h v;
#pragma unroll
  for (int i = 0; i < 16; ++i) {
    int k = ks * 32 + hsel * 8 + ((i < 8) ? i : (i + 8));
    v[i] = (_Float16)W[(size_t)k * FDIM + n];
  }
  *(v16h*)(wp + (size_t)t * FRAG_HALVES) = v;
}

// -------------------------- WMMA GEMM: H = A @ W ---------------------------
// One block per 16-row M tile; 8 waves x 2 col-tiles = 256 output columns.
// Inner loop: 3 contiguous 32B fragment loads + 2 v_wmma per K-step.
__global__ __launch_bounds__(256)
void gat_gemm_wmma(const _Float16* __restrict__ Ap,
                   const _Float16* __restrict__ Wp,
                   float* __restrict__ Hout, int M) {
  const int lane  = threadIdx.x & 31;
  const int wave  = threadIdx.x >> 5;              // 0..7
  const int mtile = blockIdx.x;
  const int rlow  = lane & 15;
  const int hsel  = lane >> 4;

  const _Float16* abase =
      Ap + ((size_t)mtile * KSTEPS * 32 + lane) * FRAG_HALVES;
  const int nt0 = wave * 2;
  const int nt1 = wave * 2 + 1;
  const _Float16* b0base = Wp + ((size_t)nt0 * KSTEPS * 32 + lane) * FRAG_HALVES;
  const _Float16* b1base = Wp + ((size_t)nt1 * KSTEPS * 32 + lane) * FRAG_HALVES;

  // Warm the packed-B stream (global_prefetch_b8).
  __builtin_prefetch(b0base, 0, 1);
  __builtin_prefetch(b1base, 0, 1);

  v8f acc0 = {};
  v8f acc1 = {};
  const size_t kstride = (size_t)32 * FRAG_HALVES; // halves per K-step

#pragma unroll
  for (int ks = 0; ks < KSTEPS; ++ks) {
    v16h a  = *(const v16h*)(abase  + (size_t)ks * kstride);
    v16h b0 = *(const v16h*)(b0base + (size_t)ks * kstride);
    v16h b1 = *(const v16h*)(b1base + (size_t)ks * kstride);
    acc0 = __builtin_amdgcn_wmma_f32_16x16x32_f16(false, a, false, b0,
                                                  (short)0, acc0, false, false);
    acc1 = __builtin_amdgcn_wmma_f32_16x16x32_f16(false, a, false, b1,
                                                  (short)0, acc1, false, false);
  }

  // C/D layout: VGPR j -> row = hsel*8 + j, col = lane&15
  const int n0 = nt0 * 16 + rlow;
  const int n1 = nt1 * 16 + rlow;
#pragma unroll
  for (int j = 0; j < 8; ++j) {
    int rm = mtile * 16 + hsel * 8 + j;
    if (rm < M) {
      Hout[(size_t)rm * FDIM + n0] = acc0[j];
      Hout[(size_t)rm * FDIM + n1] = acc1[j];
    }
  }
}

// ------------------- attention logits per (node, head) ---------------------
__global__ void gat_coeff(const float* __restrict__ Hf,
                          const float* __restrict__ att_s,
                          const float* __restrict__ att_d,
                          float* __restrict__ a_s, float* __restrict__ a_d) {
  int t = blockIdx.x * blockDim.x + threadIdx.x;      // n*HEADS + h
  if (t >= NNODES * HEADS) return;
  int n = t >> 2, hd = t & 3;
  const float4* hp = (const float4*)(Hf + (size_t)n * FDIM + hd * CH);
  const float4* sp = (const float4*)(att_s + hd * CH);
  const float4* dp = (const float4*)(att_d + hd * CH);
  float ss = 0.f, sd = 0.f;
#pragma unroll 4
  for (int c = 0; c < CH / 4; ++c) {
    float4 v = hp[c], s4 = sp[c], d4 = dp[c];
    ss += v.x * s4.x + v.y * s4.y + v.z * s4.z + v.w * s4.w;
    sd += v.x * d4.x + v.y * d4.y + v.z * d4.z + v.w * d4.w;
  }
  a_s[t] = ss;
  a_d[t] = sd;
}

// Monotone float <-> uint bijection for atomic float max via u32 max.
__device__ __forceinline__ unsigned fmap(float x) {
  unsigned u = __float_as_uint(x);
  return (u & 0x80000000u) ? ~u : (u | 0x80000000u);
}
__device__ __forceinline__ float funmap(unsigned v) {
  return (v & 0x80000000u) ? __uint_as_float(v ^ 0x80000000u)
                           : __uint_as_float(~v);
}

__device__ __forceinline__ void edge_sd(const int* __restrict__ ei, int e,
                                        int& s, int& d) {
  if (e < NEDGES) { s = ei[e]; d = ei[NEDGES + e]; }
  else            { s = e - NEDGES; d = s; }          // self loop
}

// ------------------------ segment max of edge logits -----------------------
__global__ void gat_edge_max(const int* __restrict__ ei,
                             const float* __restrict__ a_s,
                             const float* __restrict__ a_d,
                             unsigned* __restrict__ mmax) {
  int t = blockIdx.x * blockDim.x + threadIdx.x;      // e*HEADS + h
  if (t >= ETOT * HEADS) return;
  int e = t >> 2, hd = t & 3, s, d;
  edge_sd(ei, e, s, d);
  float v = a_s[s * HEADS + hd] + a_d[d * HEADS + hd];
  v = (v > 0.f) ? v : NEG_SLOPE * v;                  // leaky relu
  atomicMax(&mmax[d * HEADS + hd], fmap(v));
}

// ------------------- exp(e - max) and segment denominator ------------------
__global__ void gat_edge_exp(const int* __restrict__ ei,
                             const float* __restrict__ a_s,
                             const float* __restrict__ a_d,
                             const unsigned* __restrict__ mmax,
                             float* __restrict__ denom,
                             float* __restrict__ ealpha) {
  int t = blockIdx.x * blockDim.x + threadIdx.x;
  if (t >= ETOT * HEADS) return;
  int e = t >> 2, hd = t & 3, s, d;
  edge_sd(ei, e, s, d);
  float v = a_s[s * HEADS + hd] + a_d[d * HEADS + hd];
  v = (v > 0.f) ? v : NEG_SLOPE * v;
  float ex = __expf(v - funmap(mmax[d * HEADS + hd]));
  ealpha[t] = ex;
  atomicAdd(&denom[d * HEADS + hd], ex);
}

// ------------------ alpha-weighted message scatter-add ---------------------
__global__ void gat_edge_agg(const int* __restrict__ ei,
                             const float* __restrict__ Hf,
                             const float* __restrict__ ealpha,
                             const float* __restrict__ denom,
                             float* __restrict__ agg) {
  int t = blockIdx.x * blockDim.x + threadIdx.x;      // e*HEADS + h
  if (t >= ETOT * HEADS) return;
  int e = t >> 2, hd = t & 3, s, d;
  edge_sd(ei, e, s, d);
  float alpha = ealpha[t] / (denom[d * HEADS + hd] + 1e-16f);
  const float4* hp = (const float4*)(Hf + (size_t)s * FDIM + hd * CH);
  float*        op = agg + (size_t)d * FDIM + hd * CH;
#pragma unroll 4
  for (int c = 0; c < CH / 4; ++c) {
    float4 v = hp[c];
    atomicAdd(&op[4 * c + 0], v.x * alpha);
    atomicAdd(&op[4 * c + 1], v.y * alpha);
    atomicAdd(&op[4 * c + 2], v.z * alpha);
    atomicAdd(&op[4 * c + 3], v.w * alpha);
  }
}

// --------------------------- bias + ELU epilogue ---------------------------
__global__ void gat_bias_elu(const float* __restrict__ agg,
                             const float* __restrict__ bias,
                             float* __restrict__ out) {
  int t = blockIdx.x * blockDim.x + threadIdx.x;
  if (t >= NNODES * FDIM) return;
  float v = agg[t] + bias[t & (FDIM - 1)];
  out[t] = (v > 0.f) ? v : (__expf(v) - 1.f);
}

// ---------------------------------------------------------------------------

static void run_gat_layer(const float* act_in, const float* W,
                          const float* att_s, const float* att_d,
                          const float* bias, const int* ei,
                          _Float16* apack, _Float16* wpack,
                          float* h, float* agg, float* a_s, float* a_d,
                          unsigned* mmax, float* denom, float* ealpha,
                          float* act_out, hipStream_t stream) {
  hipMemsetAsync(agg,   0, (size_t)NNODES * FDIM * sizeof(float), stream);
  hipMemsetAsync(mmax,  0, (size_t)NNODES * HEADS * sizeof(unsigned), stream);
  hipMemsetAsync(denom, 0, (size_t)NNODES * HEADS * sizeof(float), stream);

  int na = MTILES * KSTEPS * 32;
  int nw = NTILES * KSTEPS * 32;
  gat_pack_a<<<(na + 255) / 256, 256, 0, stream>>>(act_in, apack, NNODES);
  gat_pack_w<<<(nw + 255) / 256, 256, 0, stream>>>(W, wpack);

  gat_gemm_wmma<<<MTILES, 256, 0, stream>>>(apack, wpack, h, NNODES);

  int nh = NNODES * HEADS;
  gat_coeff<<<(nh + 255) / 256, 256, 0, stream>>>(h, att_s, att_d, a_s, a_d);

  int eh = ETOT * HEADS;
  gat_edge_max<<<(eh + 255) / 256, 256, 0, stream>>>(ei, a_s, a_d, mmax);
  gat_edge_exp<<<(eh + 255) / 256, 256, 0, stream>>>(ei, a_s, a_d, mmax,
                                                     denom, ealpha);
  gat_edge_agg<<<(eh + 255) / 256, 256, 0, stream>>>(ei, h, ealpha, denom, agg);

  int nf = NNODES * FDIM;
  gat_bias_elu<<<(nf + 255) / 256, 256, 0, stream>>>(agg, bias, act_out);
}

extern "C" void kernel_launch(void* const* d_in, const int* in_sizes, int n_in,
                              void* d_out, int out_size, void* d_ws, size_t ws_size,
                              hipStream_t stream) {
  // setup_inputs order:
  // 0:x 1:edge_index 2:W1 3:a1s 4:a1d 5:b1 6:W2 7:a2s 8:a2d 9:b2
  // 10:W3 11:a3s 12:a3d 13:b3
  const float* x   = (const float*)d_in[0];
  const int*   ei  = (const int*)d_in[1];             // integer -> int32 per contract
  const float* W1  = (const float*)d_in[2];
  const float* a1s = (const float*)d_in[3];
  const float* a1d = (const float*)d_in[4];
  const float* b1  = (const float*)d_in[5];
  const float* W2  = (const float*)d_in[6];
  const float* a2s = (const float*)d_in[7];
  const float* a2d = (const float*)d_in[8];
  const float* b2  = (const float*)d_in[9];
  const float* W3  = (const float*)d_in[10];
  const float* a3s = (const float*)d_in[11];
  const float* a3d = (const float*)d_in[12];
  const float* b3  = (const float*)d_in[13];
  float* out = (float*)d_out;

  // Workspace carve-up (256B aligned slices of d_ws).
  char* ws = (char*)d_ws;
  size_t off = 0;
  auto carve = [&](size_t bytes) -> char* {
    char* p = ws + off;
    off = (off + bytes + 255) & ~(size_t)255;
    return p;
  };
  float*     h      = (float*)    carve((size_t)NNODES * FDIM * sizeof(float));
  float*     agg    = (float*)    carve((size_t)NNODES * FDIM * sizeof(float));
  float*     actA   = (float*)    carve((size_t)NNODES * FDIM * sizeof(float));
  float*     actB   = (float*)    carve((size_t)NNODES * FDIM * sizeof(float));
  _Float16*  apack  = (_Float16*) carve((size_t)MTILES * KSTEPS * 32 *
                                        FRAG_HALVES * sizeof(_Float16));
  _Float16*  wpack  = (_Float16*) carve((size_t)NTILES * KSTEPS * 32 *
                                        FRAG_HALVES * sizeof(_Float16));
  float*     a_s    = (float*)    carve((size_t)NNODES * HEADS * sizeof(float));
  float*     a_d    = (float*)    carve((size_t)NNODES * HEADS * sizeof(float));
  unsigned*  mmax   = (unsigned*) carve((size_t)NNODES * HEADS * sizeof(unsigned));
  float*     denom  = (float*)    carve((size_t)NNODES * HEADS * sizeof(float));
  float*     ealpha = (float*)    carve((size_t)ETOT * HEADS * sizeof(float));
  (void)ws_size; (void)in_sizes; (void)n_in; (void)out_size;

  // Layer 1: x -> actA ; Layer 2: actA -> actB ; Layer 3: actB -> out
  run_gat_layer(x,    W1, a1s, a1d, b1, ei, apack, wpack, h, agg, a_s, a_d,
                mmax, denom, ealpha, actA, stream);
  run_gat_layer(actA, W2, a2s, a2d, b2, ei, apack, wpack, h, agg, a_s, a_d,
                mmax, denom, ealpha, actB, stream);
  run_gat_layer(actB, W3, a3s, a3d, b3, ei, apack, wpack, h, agg, a_s, a_d,
                mmax, denom, ealpha, out, stream);
}
